// FeatExtractorLSTM_23768349016502
// MI455X (gfx1250) — matrix operand order repaired and checked by reading, not compiled, not run on hardware
//
#include <hip/hip_runtime.h>

typedef __attribute__((ext_vector_type(16))) _Float16 v16h;
typedef __attribute__((ext_vector_type(8)))  _Float16 v8h;
typedef __attribute__((ext_vector_type(8)))  float    v8f;

#define BB 256
#define TT 2048
#define II 128
#define HH 128

__device__ __forceinline__ float sigmoid_fast(float x) {
    return 1.0f / (1.0f + __expf(-x));
}
__device__ __forceinline__ float tanh_fast(float x) {
    x = fminf(fmaxf(x, -15.0f), 15.0f);   // avoid exp overflow; tanh saturated anyway
    float e = __expf(2.0f * x);
    return (e - 1.0f) / (e + 1.0f);
}

__global__ __launch_bounds__(256, 1)
void lstm_tail_kernel(const float* __restrict__ x,
                      const int*   __restrict__ lengths,
                      const float* __restrict__ W_ih,
                      const float* __restrict__ W_hh,
                      const float* __restrict__ b_ih,
                      const float* __restrict__ b_hh,
                      float* __restrict__ out)
{
    __shared__ _Float16 h_lds[16][HH];        // 4 KB, current hidden state (f16)
    __shared__ _Float16 x_lds[2][16][II];     // 8 KB, double-buffered x(t) tile (f16)
    __shared__ int s_any;

    const int tid     = threadIdx.x;
    const int lane    = tid & 31;
    const int wv      = tid >> 5;             // wave 0..7 -> owns h-col tile wv
    const int rowbase = blockIdx.x * 16;

    // ---- early exit: output is zero unless some row has full length ----
    if (tid == 0) {
        int a = 0;
        for (int r = 0; r < 16; ++r) a |= (lengths[rowbase + r] == TT) ? 1 : 0;
        s_any = a;
    }
    __syncthreads();
    if (!s_any) {
        for (int k = tid; k < 16 * HH; k += 256) out[(size_t)rowbase * HH + k] = 0.0f;
        return;
    }

    // ---- build register-resident B fragments: B[k][n] = W[n][k] (f32 -> f16) ----
    const int nlo  = lane & 15;               // column within 16-wide tile
    const int kgrp = (lane >> 4) * 8;         // B layout: lanes 16-31 hold K=8..15,24..31
    v16h Bh[4][4], Bx[4][4];                  // [gate][k-chunk], K-chunk = 32
    float bias[4];
    #pragma unroll
    for (int g = 0; g < 4; ++g) {
        const int n = g * 128 + wv * 16 + nlo;         // gate column 0..511
        bias[g] = b_ih[n] + b_hh[n];
        #pragma unroll
        for (int kc = 0; kc < 4; ++kc) {
            const int ks = kc * 32 + kgrp;
            const float* ph = W_hh + (size_t)n * HH + ks;
            const float* px = W_ih + (size_t)n * II + ks;
            v16h bh, bx;
            #pragma unroll
            for (int e = 0; e < 16; ++e) {
                bh[e] = (_Float16)ph[e < 8 ? e : e + 8];
                bx[e] = (_Float16)px[e < 8 ? e : e + 8];
            }
            Bh[g][kc] = bh;
            Bx[g][kc] = bx;
        }
    }

    // ---- init h=0 in LDS, stage x(t=0) ----
    for (int k = tid; k < 16 * HH; k += 256) ((_Float16*)h_lds)[k] = (_Float16)0.0f;
    const int xr = tid >> 4;                  // row 0..15
    const int xc = (tid & 15) * 8;            // col 0..120
    {
        const float* px = x + ((size_t)(rowbase + xr) * TT) * II + xc;
        #pragma unroll
        for (int e = 0; e < 8; ++e) x_lds[0][xr][xc + e] = (_Float16)px[e];
    }
    __syncthreads();

    // per-lane state tiles (C/D layout: VGPR r -> row M0+r, lane -> col)
    const int M0 = (lane >> 4) * 8;
    const int Nc = wv * 16 + (lane & 15);
    float cst[8], hst[8];
    #pragma unroll
    for (int r = 0; r < 8; ++r) { cst[r] = 0.0f; hst[r] = 0.0f; }

    const int arow  = lane & 15;              // A layout: lane 16-31 -> same rows
    const int akoff = (lane >> 4) * 8;        // A layout K offset

    int cur = 0;
    for (int t = 0; t < TT; ++t) {
        // prefetch next x tile into registers (hidden under WMMA chain)
        float xf[8];
        if (t + 1 < TT) {
            const float* px = x + ((size_t)(rowbase + xr) * TT + (t + 1)) * II + xc;
            #pragma unroll
            for (int e = 0; e < 8; ++e) xf[e] = px[e];
        }

        v8f acc[4];
        #pragma unroll
        for (int g = 0; g < 4; ++g) {
            #pragma unroll
            for (int e = 0; e < 8; ++e) acc[g][e] = bias[g];
        }

        // gates += h @ W_hh^T   (4 K-chunks x 4 gate tiles)
        #pragma unroll
        for (int kc = 0; kc < 4; ++kc) {
            const v8h lo = *(const v8h*)&h_lds[arow][kc * 32 + akoff];
            const v8h hi = *(const v8h*)&h_lds[arow][kc * 32 + akoff + 16];
            v16h a;
            #pragma unroll
            for (int e = 0; e < 8; ++e) { a[e] = lo[e]; a[8 + e] = hi[e]; }
            #pragma unroll
            for (int g = 0; g < 4; ++g)
                acc[g] = __builtin_amdgcn_wmma_f32_16x16x32_f16(
                    false, a, false, Bh[g][kc], (short)0, acc[g], false, false);
        }
        // gates += x(t) @ W_ih^T
        #pragma unroll
        for (int kc = 0; kc < 4; ++kc) {
            const v8h lo = *(const v8h*)&x_lds[cur][arow][kc * 32 + akoff];
            const v8h hi = *(const v8h*)&x_lds[cur][arow][kc * 32 + akoff + 16];
            v16h a;
            #pragma unroll
            for (int e = 0; e < 8; ++e) { a[e] = lo[e]; a[8 + e] = hi[e]; }
            #pragma unroll
            for (int g = 0; g < 4; ++g)
                acc[g] = __builtin_amdgcn_wmma_f32_16x16x32_f16(
                    false, a, false, Bx[g][kc], (short)0, acc[g], false, false);
        }

        // elementwise gate math (i, f, g, o)
        #pragma unroll
        for (int r = 0; r < 8; ++r) {
            const float iv = sigmoid_fast(acc[0][r]);
            const float fv = sigmoid_fast(acc[1][r]);
            const float gv = tanh_fast(acc[2][r]);
            const float ov = sigmoid_fast(acc[3][r]);
            const float cn = fv * cst[r] + iv * gv;
            cst[r] = cn;
            hst[r] = ov * tanh_fast(cn);
        }

        __syncthreads();                      // all A-reads of h_lds/x_lds done

        // publish new h (f16) and next x tile
        #pragma unroll
        for (int r = 0; r < 8; ++r) h_lds[M0 + r][Nc] = (_Float16)hst[r];
        if (t + 1 < TT) {
            #pragma unroll
            for (int e = 0; e < 8; ++e) x_lds[cur ^ 1][xr][xc + e] = (_Float16)xf[e];
        }
        __syncthreads();
        cur ^= 1;
    }

    // ---- final store: zero unless full-length row ----
    #pragma unroll
    for (int r = 0; r < 8; ++r) {
        const int row = rowbase + M0 + r;
        const float v = (lengths[row] == TT) ? hst[r] : 0.0f;
        out[(size_t)row * HH + Nc] = v;
    }
}

extern "C" void kernel_launch(void* const* d_in, const int* in_sizes, int n_in,
                              void* d_out, int out_size, void* d_ws, size_t ws_size,
                              hipStream_t stream) {
    const float* x       = (const float*)d_in[0];
    const int*   lengths = (const int*)d_in[1];
    const float* W_ih    = (const float*)d_in[2];
    const float* W_hh    = (const float*)d_in[3];
    const float* b_ih    = (const float*)d_in[4];
    const float* b_hh    = (const float*)d_in[5];
    float* out = (float*)d_out;

    lstm_tail_kernel<<<dim3(BB / 16), dim3(256), 0, stream>>>(
        x, lengths, W_ih, W_hh, b_ih, b_hh, out);
}
